// GAT_delta_53223234732625
// MI455X (gfx1250) — matrix operand order repaired and checked by reading, not compiled, not run on hardware
//
#include <hip/hip_runtime.h>

// ---------------------------------------------------------------------------
// GAT 3-layer forward on MI455X (gfx1250, wave32).
// Dense projections use v_wmma_f32_16x16x32_bf16 (f32 accumulate), software-
// pipelined (next K-step fragments prefetched during current WMMAs).
// Sparse edge phase: coalesced wave-per-edge streaming with f32 atomics.
// ---------------------------------------------------------------------------

#define NN 100000          // nodes
#define NE 1600000         // edges (before self loops)
#define NT_EDGES (NE + NN) // edges incl. self loops

typedef __bf16 bf16;
typedef bf16  v16bf __attribute__((ext_vector_type(16)));
typedef bf16  v8bf  __attribute__((ext_vector_type(8)));
typedef float v8f   __attribute__((ext_vector_type(8)));

// ---------------------------- small utility kernels ------------------------

__global__ void fill_f32_kernel(float* __restrict__ p, float v, int n) {
  int i = blockIdx.x * blockDim.x + threadIdx.x;
  if (i < n) p[i] = v;
}

__global__ void cvt_f32_to_bf16_kernel(const float* __restrict__ in,
                                       bf16* __restrict__ out, int n) {
  int i = blockIdx.x * blockDim.x + threadIdx.x;
  if (i < n) out[i] = (bf16)in[i];
}

// W [K, Nreal] f32 (row-major, as in reference) -> Wt [Npad, K] bf16
// (transposed, zero-padded columns) so each WMMA B-fragment is a contiguous
// 16-element bf16 run per lane.
__global__ void w_transpose_kernel(const float* __restrict__ W,
                                   bf16* __restrict__ Wt,
                                   int K, int Nreal, int Npad) {
  int i = blockIdx.x * blockDim.x + threadIdx.x;
  if (i >= Npad * K) return;
  int n = i / K, k = i % K;
  float v = (n < Nreal) ? W[(long long)k * Nreal + n] : 0.0f;
  Wt[i] = (bf16)v;
}

// ---------------------------- WMMA GEMM ------------------------------------

// two contiguous 8-element bf16 runs -> one 16-element fragment
__device__ __forceinline__ v16bf load_frag16(const bf16* __restrict__ p0,
                                             const bf16* __restrict__ p1) {
  v8bf lo = *reinterpret_cast<const v8bf*>(p0);
  v8bf hi = *reinterpret_cast<const v8bf*>(p1);
  v16bf f;
#pragma unroll
  for (int i = 0; i < 8; ++i) { f[i] = lo[i]; f[8 + i] = hi[i]; }
  return f;
}

// C[M, Ncols] (f32) = A[M, K] (bf16, row-major) x Bt[Ncols, K]^T (bf16).
// One wave computes a 16x64 tile: 4 accumulators sharing one A fragment.
// Software pipeline: K-step k+1 fragments are fetched before the K-step k
// WMMAs issue, so the matrix pipe is not stalled on loadcnt==0 per WMMA.
// M must be a multiple of 16 (100000 = 6250*16), Ncols a multiple of 64.
__global__ __launch_bounds__(256)
void gemm_bf16_wmma_kernel(const bf16* __restrict__ A,
                           const bf16* __restrict__ Bt,
                           float* __restrict__ C,
                           int M, int K, int Ncols) {
  const int lane  = threadIdx.x & 31;
  const int wave  = threadIdx.x >> 5;
  const int mtile = blockIdx.x * 8 + wave;
  if (mtile * 16 >= M) return;              // uniform per wave: EXEC stays full
  const int mbase = mtile * 16;
  const int nbase = blockIdx.y * 64;
  const int half  = lane >> 4;              // K-half select (this code's order: k = 16*half + s)
  const int l16   = lane & 15;

  // per-lane base pointers (K offset added per step)
  // A 16x32 frag: row = lane&15; elems 0-15 contiguous at K=k0+half*16.
  const bf16* aptr = A + (long long)(mbase + l16) * K + half * 16;
  // B 32x16 frag: col = lane&15; elems 0-15 contiguous at K=k0+half*16.
  const bf16* bptr0 = Bt + (long long)(nbase + 0 * 16 + l16) * K + half * 16;
  const bf16* bptr1 = Bt + (long long)(nbase + 1 * 16 + l16) * K + half * 16;
  const bf16* bptr2 = Bt + (long long)(nbase + 2 * 16 + l16) * K + half * 16;
  const bf16* bptr3 = Bt + (long long)(nbase + 3 * 16 + l16) * K + half * 16;

  v8f acc0 = {0.f,0.f,0.f,0.f,0.f,0.f,0.f,0.f};
  v8f acc1 = acc0, acc2 = acc0, acc3 = acc0;

  // prologue: fetch K-step 0
  v16bf aCur  = load_frag16(aptr, aptr + 8);
  v16bf bCur0 = load_frag16(bptr0, bptr0 + 8);
  v16bf bCur1 = load_frag16(bptr1, bptr1 + 8);
  v16bf bCur2 = load_frag16(bptr2, bptr2 + 8);
  v16bf bCur3 = load_frag16(bptr3, bptr3 + 8);

  int k0 = 0;
  for (; k0 + 32 < K; k0 += 32) {
    const int kn = k0 + 32;
    // prefetch next K-step fragments (loads issue before the WMMAs below)
    v16bf aNxt  = load_frag16(aptr  + kn, aptr  + kn + 8);
    v16bf bNxt0 = load_frag16(bptr0 + kn, bptr0 + kn + 8);
    v16bf bNxt1 = load_frag16(bptr1 + kn, bptr1 + kn + 8);
    v16bf bNxt2 = load_frag16(bptr2 + kn, bptr2 + kn + 8);
    v16bf bNxt3 = load_frag16(bptr3 + kn, bptr3 + kn + 8);

    acc0 = __builtin_amdgcn_wmma_f32_16x16x32_bf16(false, aCur, false, bCur0,
                                                   (short)0, acc0, false, false);
    acc1 = __builtin_amdgcn_wmma_f32_16x16x32_bf16(false, aCur, false, bCur1,
                                                   (short)0, acc1, false, false);
    acc2 = __builtin_amdgcn_wmma_f32_16x16x32_bf16(false, aCur, false, bCur2,
                                                   (short)0, acc2, false, false);
    acc3 = __builtin_amdgcn_wmma_f32_16x16x32_bf16(false, aCur, false, bCur3,
                                                   (short)0, acc3, false, false);

    aCur = aNxt;
    bCur0 = bNxt0; bCur1 = bNxt1; bCur2 = bNxt2; bCur3 = bNxt3;
  }
  // epilogue K-step
  acc0 = __builtin_amdgcn_wmma_f32_16x16x32_bf16(false, aCur, false, bCur0,
                                                 (short)0, acc0, false, false);
  acc1 = __builtin_amdgcn_wmma_f32_16x16x32_bf16(false, aCur, false, bCur1,
                                                 (short)0, acc1, false, false);
  acc2 = __builtin_amdgcn_wmma_f32_16x16x32_bf16(false, aCur, false, bCur2,
                                                 (short)0, acc2, false, false);
  acc3 = __builtin_amdgcn_wmma_f32_16x16x32_bf16(false, aCur, false, bCur3,
                                                 (short)0, acc3, false, false);

  // C/D layout: VGPR j -> row j + 8*(lane/16), col = lane&15.
  v8f accs[4] = {acc0, acc1, acc2, acc3};
#pragma unroll
  for (int nt = 0; nt < 4; ++nt) {
    const int col = nbase + nt * 16 + l16;
#pragma unroll
    for (int j = 0; j < 8; ++j) {
      const int row = mbase + j + half * 8;
      C[(long long)row * Ncols + col] = accs[nt][j];
    }
  }
}

// ---------------------------- attention logits ------------------------------
// al_src[n,h] = sum_c x[n, h*C+c] * a_src[h,c]; same for al_dst.
__global__ void compute_al_kernel(const float* __restrict__ x,
                                  const float* __restrict__ aS,
                                  const float* __restrict__ aD,
                                  float* __restrict__ alS,
                                  float* __restrict__ alD,
                                  int Nn, int H, int Cc, int stride) {
  int i = blockIdx.x * blockDim.x + threadIdx.x;
  if (i >= Nn * H) return;
  int n = i / H, h = i % H;
  const float* xr = x + (long long)n * stride + h * Cc;
  const float* as = aS + h * Cc;
  const float* ad = aD + h * Cc;
  float ss = 0.f, sd = 0.f;
  for (int c = 0; c < Cc; ++c) { float v = xr[c]; ss += v * as[c]; sd += v * ad[c]; }
  alS[i] = ss;
  alD[i] = sd;
}

// ---------------------------- edge phase -----------------------------------

__device__ inline void atomicMaxF32(float* addr, float value) {
  if (value >= 0.0f)
    atomicMax(reinterpret_cast<int*>(addr), __float_as_int(value));
  else
    atomicMin(reinterpret_cast<unsigned int*>(addr), __float_as_uint(value));
}

__device__ inline void edge_endpoints(const long long* __restrict__ src,
                                      const long long* __restrict__ dst,
                                      int e, int& s, int& d) {
  if (e < NE) { s = (int)src[e]; d = (int)dst[e]; }
  else        { s = d = e - NE; }     // appended self loops
}

// pass 1: m[d,h] = max over incoming edges of leaky_relu(al_src[s]+al_dst[d])
__global__ void edge_max_kernel(const long long* __restrict__ src,
                                const long long* __restrict__ dst,
                                const float* __restrict__ alS,
                                const float* __restrict__ alD,
                                float* __restrict__ m, int H) {
  int i = blockIdx.x * blockDim.x + threadIdx.x;
  if (i >= NT_EDGES * H) return;
  int e = i / H, h = i % H;
  int s, d; edge_endpoints(src, dst, e, s, d);
  float v = alS[(long long)s * H + h] + alD[(long long)d * H + h];
  v = (v > 0.f) ? v : 0.2f * v;       // leaky_relu, slope 0.2
  atomicMaxF32(&m[(long long)d * H + h], v);
}

// pass 2: w = exp(e - m[d]); s[d] += w; store per-edge w
__global__ void edge_exp_sum_kernel(const long long* __restrict__ src,
                                    const long long* __restrict__ dst,
                                    const float* __restrict__ alS,
                                    const float* __restrict__ alD,
                                    const float* __restrict__ m,
                                    float* __restrict__ wedge,
                                    float* __restrict__ sbuf, int H) {
  int i = blockIdx.x * blockDim.x + threadIdx.x;
  if (i >= NT_EDGES * H) return;
  int e = i / H, h = i % H;
  int s, d; edge_endpoints(src, dst, e, s, d);
  float v = alS[(long long)s * H + h] + alD[(long long)d * H + h];
  v = (v > 0.f) ? v : 0.2f * v;
  float w = __expf(v - m[(long long)d * H + h]);
  wedge[(long long)e * H + h] = w;
  atomicAdd(&sbuf[(long long)d * H + h], w);
}

// pass 3: out[d] += x[s] * alpha, one wave per edge, coalesced over H*C cols.
__global__ __launch_bounds__(256)
void edge_agg_kernel(const long long* __restrict__ src,
                     const long long* __restrict__ dst,
                     const float* __restrict__ x,
                     const float* __restrict__ wedge,
                     const float* __restrict__ sbuf,
                     float* __restrict__ out,
                     int H, int Cc, int stride) {
  int e    = blockIdx.x * 8 + (threadIdx.x >> 5);
  int lane = threadIdx.x & 31;
  if (e >= NT_EDGES) return;
  int s, d; edge_endpoints(src, dst, e, s, d);
  const int HC = H * Cc;
  const float* xs = x + (long long)s * stride;
  float* od       = out + (long long)d * stride;
  for (int c = lane; c < HC; c += 32) {
    int h = c / Cc;
    float alpha = wedge[(long long)e * H + h] / sbuf[(long long)d * H + h];
    atomicAdd(&od[c], xs[c] * alpha);
  }
}

// ---------------------------- epilogues ------------------------------------

// h_next = elu(out + b), emitted as bf16 for the next layer's WMMA A operand.
__global__ void epilogue_elu_kernel(const float* __restrict__ out,
                                    const float* __restrict__ b,
                                    bf16* __restrict__ hn, int total, int HC) {
  int i = blockIdx.x * blockDim.x + threadIdx.x;
  if (i >= total) return;
  float v = out[i] + b[i % HC];
  v = (v > 0.f) ? v : (__expf(v) - 1.0f);   // elu(alpha=1)
  hn[i] = (bf16)v;
}

// final layer: y[n, 0:40] = out[n, 0:40 of 64] + b2  (f32)
__global__ void final_out_kernel(const float* __restrict__ out,
                                 const float* __restrict__ b,
                                 float* __restrict__ y,
                                 int Nn, int Cc, int stride) {
  int i = blockIdx.x * blockDim.x + threadIdx.x;
  if (i >= Nn * Cc) return;
  int n = i / Cc, c = i % Cc;
  y[i] = out[(long long)n * stride + c] + b[c];
}

// ---------------------------- host orchestration ---------------------------

static inline int cdiv(long long a, long long b) { return (int)((a + b - 1) / b); }

extern "C" void kernel_launch(void* const* d_in, const int* in_sizes, int n_in,
                              void* d_out, int out_size, void* d_ws, size_t ws_size,
                              hipStream_t stream) {
  (void)in_sizes; (void)n_in; (void)out_size; (void)ws_size;

  const float*     feats = (const float*)d_in[0];
  const long long* eidx  = (const long long*)d_in[1];   // [2, NE] int64
  const long long* esrc  = eidx;
  const long long* edst  = eidx + NE;
  const float* W0 = (const float*)d_in[2],  *aS0 = (const float*)d_in[3],
             *aD0 = (const float*)d_in[4],  *b0  = (const float*)d_in[5];
  const float* W1 = (const float*)d_in[6],  *aS1 = (const float*)d_in[7],
             *aD1 = (const float*)d_in[8],  *b1  = (const float*)d_in[9];
  const float* W2 = (const float*)d_in[10], *aS2 = (const float*)d_in[11],
             *aD2 = (const float*)d_in[12], *b2  = (const float*)d_in[13];

  // workspace carve-up (256B aligned)
  char* ws = (char*)d_ws;
  size_t off = 0;
  auto carve = [&](size_t bytes) -> void* {
    void* p = ws + off;
    off += (bytes + 255) & ~(size_t)255;
    return p;
  };
  float* xbuf   = (float*)carve((size_t)NN * 256 * 4);       // projected features
  float* outbuf = (float*)carve((size_t)NN * 256 * 4);       // aggregation target
  bf16*  hbf    = (bf16*) carve((size_t)NN * 256 * 2);       // layer input, bf16
  float* wedge  = (float*)carve((size_t)NT_EDGES * 8 * 4);   // per-edge weights
  float* alS    = (float*)carve((size_t)NN * 8 * 4);
  float* alD    = (float*)carve((size_t)NN * 8 * 4);
  float* mbuf   = (float*)carve((size_t)NN * 8 * 4);
  float* sbuf   = (float*)carve((size_t)NN * 8 * 4);
  bf16*  W0t    = (bf16*) carve((size_t)256 * 128 * 2);      // [256][128]
  bf16*  W1t    = (bf16*) carve((size_t)256 * 256 * 2);      // [256][256]
  bf16*  W2t    = (bf16*) carve((size_t)64  * 256 * 2);      // [64][256] (40 real)

  const int TB = 256;

  // input + weight conversion
  cvt_f32_to_bf16_kernel<<<cdiv((long long)NN * 128, TB), TB, 0, stream>>>(
      feats, hbf, NN * 128);
  w_transpose_kernel<<<cdiv(256 * 128, TB), TB, 0, stream>>>(W0, W0t, 128, 256, 256);
  w_transpose_kernel<<<cdiv(256 * 256, TB), TB, 0, stream>>>(W1, W1t, 256, 256, 256);
  w_transpose_kernel<<<cdiv(64  * 256, TB), TB, 0, stream>>>(W2, W2t, 256, 40, 64);

  struct Layer {
    const bf16* Wt; const float *aS, *aD, *b;
    int K, H, C, stride;   // stride = padded output cols
  };
  const Layer layers[3] = {
      {W0t, aS0, aD0, b0, 128, 8, 32, 256},
      {W1t, aS1, aD1, b1, 256, 8, 32, 256},
      {W2t, aS2, aD2, b2, 256, 1, 40, 64},
  };

  const int mblocks = cdiv((long long)NN / 16, 8);   // 8 waves (m-tiles) / block

  for (int L = 0; L < 3; ++L) {
    const Layer& P = layers[L];
    const int NH = NN * P.H;

    // 1) x = h @ W  (WMMA)
    dim3 ggrid(mblocks, P.stride / 64);
    gemm_bf16_wmma_kernel<<<ggrid, TB, 0, stream>>>(hbf, P.Wt, xbuf, NN, P.K, P.stride);

    // 2) attention logits
    compute_al_kernel<<<cdiv(NH, TB), TB, 0, stream>>>(
        xbuf, P.aS, P.aD, alS, alD, NN, P.H, P.C, P.stride);

    // 3) segment max / softmax denominators / zero aggregation buffer
    fill_f32_kernel<<<cdiv(NH, TB), TB, 0, stream>>>(mbuf, -3.0e38f, NH);
    fill_f32_kernel<<<cdiv(NH, TB), TB, 0, stream>>>(sbuf, 0.0f, NH);
    fill_f32_kernel<<<cdiv((long long)NN * P.stride, TB), TB, 0, stream>>>(
        outbuf, 0.0f, NN * P.stride);

    edge_max_kernel<<<cdiv((long long)NT_EDGES * P.H, TB), TB, 0, stream>>>(
        esrc, edst, alS, alD, mbuf, P.H);
    edge_exp_sum_kernel<<<cdiv((long long)NT_EDGES * P.H, TB), TB, 0, stream>>>(
        esrc, edst, alS, alD, mbuf, wedge, sbuf, P.H);

    // 4) weighted aggregation: wave per edge
    edge_agg_kernel<<<cdiv(NT_EDGES, 8), TB, 0, stream>>>(
        esrc, edst, xbuf, wedge, sbuf, outbuf, P.H, P.C, P.stride);

    // 5) epilogue
    if (L < 2) {
      epilogue_elu_kernel<<<cdiv((long long)NN * P.stride, TB), TB, 0, stream>>>(
          outbuf, P.b, hbf, NN * P.stride, P.stride);
    } else {
      final_out_kernel<<<cdiv((long long)NN * P.C, TB), TB, 0, stream>>>(
          outbuf, P.b, (float*)d_out, NN, P.C, P.stride);
    }
  }
}
